// Model_74062416053270
// MI455X (gfx1250) — matrix operand order repaired and checked by reading, not compiled, not run on hardware
//
#include <hip/hip_runtime.h>
#include <math.h>

#define N_TOK 8192
#define DIM   512
#define FFN   2048
#define NEXP  8
#define TILE_M 32
#define FCHUNK 512
#define HROW  (FCHUNK + 8)        // pad -> row stride 1040B, conflict-free b128 LDS reads
#define MOE_EPS 2.220446049250313e-16f

typedef __bf16  v16bf __attribute__((ext_vector_type(16)));
typedef float   v8f   __attribute__((ext_vector_type(8)));

union FragBF { v16bf v; __bf16 e[16]; uint4 q[2]; };

__device__ __forceinline__ v8f wmma_bf16(const FragBF& a, const FragBF& b, v8f c) {
  return __builtin_amdgcn_wmma_f32_16x16x32_bf16(false, a.v, false, b.v,
                                                 (short)0, c, false, false);
}

// Fragment = two contiguous 16B chunks (8 bf16 each). A: chunks at k and k+16.
// B: chunks at k and k+16 (same K order as A). Layouts per ISA 7.12.2.
__device__ __forceinline__ void load_frag(FragBF& f, const __bf16* p0, const __bf16* p1) {
  f.q[0] = *(const uint4*)(p0);
  f.q[1] = *(const uint4*)(p1);
}

// ---------------- one-time conversions ----------------
__global__ void __launch_bounds__(256) cvt_x_kernel(const float* __restrict__ in,
                                                    __bf16* __restrict__ out) {
  int i = (blockIdx.x * blockDim.x + threadIdx.x) * 8;
  float4 a = *(const float4*)(in + i);
  float4 b = *(const float4*)(in + i + 4);
  union { __bf16 e[8]; uint4 q; } u;
  u.e[0] = (__bf16)a.x; u.e[1] = (__bf16)a.y; u.e[2] = (__bf16)a.z; u.e[3] = (__bf16)a.w;
  u.e[4] = (__bf16)b.x; u.e[5] = (__bf16)b.y; u.e[6] = (__bf16)b.z; u.e[7] = (__bf16)b.w;
  *(uint4*)(out + i) = u.q;
}

// in: [E][R][C] f32  ->  out: [E][C][R] bf16   (LDS-tiled 32x32 transpose)
__global__ void __launch_bounds__(256) transpose_cvt_kernel(const float* __restrict__ in,
                                                            __bf16* __restrict__ out,
                                                            int R, int C) {
  __shared__ float tile[32][33];
  const int e = blockIdx.z;
  const float* src = in + (size_t)e * R * C;
  __bf16* dst = out + (size_t)e * R * C;
  const int c0 = blockIdx.x * 32, r0 = blockIdx.y * 32;
  const int tx = threadIdx.x, ty = threadIdx.y;
#pragma unroll
  for (int i = ty; i < 32; i += 8)
    tile[i][tx] = src[(size_t)(r0 + i) * C + c0 + tx];
  __syncthreads();
#pragma unroll
  for (int i = ty; i < 32; i += 8)
    dst[(size_t)(c0 + i) * R + r0 + tx] = (__bf16)tile[tx][i];
}

// ---------------- kernel 1: gating (top-2 softmax + routing lists) ----------------
__global__ void __launch_bounds__(256) gate_kernel(const float* __restrict__ x,
                                                   const float* __restrict__ Wg,
                                                   int* __restrict__ cnt,
                                                   int* __restrict__ route,
                                                   float* __restrict__ gate_val) {
  int n = blockIdx.x * blockDim.x + threadIdx.x;
  if (n >= N_TOK) return;
  float lg[NEXP];
#pragma unroll
  for (int e = 0; e < NEXP; ++e) lg[e] = 0.f;
  const float* xr = x + (size_t)n * DIM;
  for (int d = 0; d < DIM; ++d) {
    float xv = xr[d];
#pragma unroll
    for (int e = 0; e < NEXP; ++e) lg[e] = fmaf(xv, Wg[d * NEXP + e], lg[e]);
  }
  int i1 = 0; float v1 = lg[0];
#pragma unroll
  for (int e = 1; e < NEXP; ++e) if (lg[e] > v1) { v1 = lg[e]; i1 = e; }
  int i2 = -1; float v2 = -INFINITY;
#pragma unroll
  for (int e = 0; e < NEXP; ++e) if (e != i1 && lg[e] > v2) { v2 = lg[e]; i2 = e; }
  float g1 = 1.f / (1.f + __expf(v2 - v1));   // softmax over {v1, v2}, v1 >= v2
  float g2 = 1.f - g1;
  int p1 = atomicAdd(&cnt[i1], 1);
  route[i1 * N_TOK + p1] = n * 2;
  gate_val[n * 2] = g1;
  int p2 = atomicAdd(&cnt[i2], 1);
  route[i2 * N_TOK + p2] = n * 2 + 1;
  gate_val[n * 2 + 1] = g2;
}

// ---------------- kernel 2: per-expert MLP on routed 32-token tiles ----------------
// xbf: [N][D] bf16 ; W1t: [E][F][D] bf16 (W1^T) ; W2t: [E][D][F] bf16 (W2^T)
__global__ void __launch_bounds__(256)
__attribute__((amdgpu_waves_per_eu(4)))
moe_expert_kernel(
    const __bf16* __restrict__ xbf,
    const __bf16* __restrict__ W1t, const float* __restrict__ b1,
    const __bf16* __restrict__ W2t, const float* __restrict__ b2,
    const int* __restrict__ cnt, const int* __restrict__ route,
    const float* __restrict__ gate_val, float* __restrict__ slot_out) {
  const int e = blockIdx.y;
  const int cnt_e = cnt[e];
  const int base = blockIdx.x * TILE_M;
  if (base >= cnt_e) return;
  const int mvalid = min(TILE_M, cnt_e - base);

  __shared__ __bf16 h_lds[TILE_M][HROW];   // ~32.5 KB
  __shared__ int    s_route[TILE_M];
  __shared__ int    s_tok[TILE_M];
  __shared__ float  s_gate[TILE_M];

  const int tid = threadIdx.x;
  if (tid < TILE_M) {
    int r = route[e * N_TOK + base + ((tid < mvalid) ? tid : 0)];
    s_route[tid] = r;
    s_tok[tid]   = r >> 1;
    s_gate[tid]  = gate_val[r];
  }
  __syncthreads();

  const int w    = tid >> 5;
  const int lane = tid & 31;
  const int nlo  = lane & 15;
  const bool hih = (lane & 16) != 0;
  const int koff = hih ? 8 : 0;    // A-fragment K start within 32-chunk
  const int kbB  = hih ? 8 : 0;    // B-fragment K start within 32-chunk
  const int mof  = hih ? 8 : 0;    // C/D row offset for this half-wave

  const __bf16* W1e = W1t + (size_t)e * DIM * FFN;  // [F][D]
  const __bf16* W2e = W2t + (size_t)e * FFN * DIM;  // [D][F]
  const __bf16* xr0 = xbf + (size_t)s_tok[nlo] * DIM;
  const __bf16* xr1 = xbf + (size_t)s_tok[16 + nlo] * DIM;

  v8f acc_o[2][4];
#pragma unroll
  for (int rt = 0; rt < 2; ++rt)
#pragma unroll
    for (int t = 0; t < 4; ++t) acc_o[rt][t] = (v8f){0.f,0.f,0.f,0.f,0.f,0.f,0.f,0.f};

  for (int c = 0; c < 4; ++c) {
    const int fbase = c * FCHUNK;
    // ---- phase 1: h = relu(x @ W1 + b1) for this 512-wide F chunk ----
    v8f acc_h[2][4];
#pragma unroll
    for (int rt = 0; rt < 2; ++rt)
#pragma unroll
      for (int t = 0; t < 4; ++t) acc_h[rt][t] = (v8f){0.f,0.f,0.f,0.f,0.f,0.f,0.f,0.f};

#pragma unroll 2
    for (int kb = 0; kb < DIM; kb += 32) {
      FragBF a0, a1;
      load_frag(a0, xr0 + kb + koff, xr0 + kb + koff + 16);
      load_frag(a1, xr1 + kb + koff, xr1 + kb + koff + 16);
#pragma unroll
      for (int t = 0; t < 4; ++t) {
        const int col = fbase + w * 64 + t * 16 + nlo;
        const __bf16* bp = W1e + (size_t)col * DIM + kb + kbB;  // K chunks at kbB, kbB+16
        FragBF bf;
        load_frag(bf, bp, bp + 16);
        acc_h[0][t] = wmma_bf16(a0, bf, acc_h[0][t]);
        acc_h[1][t] = wmma_bf16(a1, bf, acc_h[1][t]);
      }
    }
    // store h tile to LDS (bf16, A-layout-compatible packed pairs)
#pragma unroll
    for (int rt = 0; rt < 2; ++rt)
#pragma unroll
      for (int t = 0; t < 4; ++t) {
        const int lc  = w * 64 + t * 16 + nlo;
        const float b1v = b1[e * FFN + fbase + lc];
#pragma unroll
        for (int j = 0; j < 8; ++j) {
          const int m = rt * 16 + j + mof;
          float v = acc_h[rt][t][j] + b1v;
          h_lds[m][lc] = (__bf16)(v > 0.f ? v : 0.f);
        }
      }
    __syncthreads();
    // ---- phase 2: o += h @ W2 for this F chunk ----
#pragma unroll 2
    for (int kb = 0; kb < FCHUNK; kb += 32) {
      FragBF a0, a1;
      load_frag(a0, &h_lds[nlo][kb + koff],      &h_lds[nlo][kb + koff + 16]);
      load_frag(a1, &h_lds[16 + nlo][kb + koff], &h_lds[16 + nlo][kb + koff + 16]);
#pragma unroll
      for (int t = 0; t < 4; ++t) {
        const int col = w * 64 + t * 16 + nlo;
        const __bf16* bp = W2e + (size_t)col * FFN + fbase + kb + kbB;  // K chunks at kbB, kbB+16
        FragBF bf;
        load_frag(bf, bp, bp + 16);
        acc_o[0][t] = wmma_bf16(a0, bf, acc_o[0][t]);
        acc_o[1][t] = wmma_bf16(a1, bf, acc_o[1][t]);
      }
    }
    __syncthreads();
  }
  // ---- finalize: slot_out[r] = gate * exp(o + b2) ----
#pragma unroll
  for (int rt = 0; rt < 2; ++rt)
#pragma unroll
    for (int t = 0; t < 4; ++t) {
      const int col = w * 64 + t * 16 + nlo;
      const float b2v = b2[e * DIM + col];
#pragma unroll
      for (int j = 0; j < 8; ++j) {
        const int m = rt * 16 + j + mof;
        if (m < mvalid) {
          const int r = s_route[m];
          slot_out[(size_t)r * DIM + col] = s_gate[m] * __expf(acc_o[rt][t][j] + b2v);
        }
      }
    }
}

// ---------------- kernel 3: combine two slots, eps floor, log ----------------
__device__ __forceinline__ float comb1(float a, float b) {
  float c = a + b;
  if (c == 0.f) c = MOE_EPS;
  return __logf(c);
}

__global__ void __launch_bounds__(256) combine_kernel(const float* __restrict__ slot_out,
                                                      float* __restrict__ out) {
  int idx = blockIdx.x * blockDim.x + threadIdx.x;     // one float4 per thread
  int n  = idx / (DIM / 4);
  int d4 = (idx % (DIM / 4)) * 4;
  const float4 s0 = *(const float4*)(slot_out + ((size_t)n * 2)     * DIM + d4);
  const float4 s1 = *(const float4*)(slot_out + ((size_t)n * 2 + 1) * DIM + d4);
  float4 o;
  o.x = comb1(s0.x, s1.x);
  o.y = comb1(s0.y, s1.y);
  o.z = comb1(s0.z, s1.z);
  o.w = comb1(s0.w, s1.w);
  *(float4*)(out + (size_t)n * DIM + d4) = o;
}

extern "C" void kernel_launch(void* const* d_in, const int* in_sizes, int n_in,
                              void* d_out, int out_size, void* d_ws, size_t ws_size,
                              hipStream_t stream) {
  const float* x  = (const float*)d_in[0];
  const float* Wg = (const float*)d_in[1];
  const float* W1 = (const float*)d_in[2];
  const float* b1 = (const float*)d_in[3];
  const float* W2 = (const float*)d_in[4];
  const float* b2 = (const float*)d_in[5];
  float* out = (float*)d_out;

  char* ws = (char*)d_ws;
  size_t off = 0;
  int*   cnt      = (int*)(ws + off);  off += 256;                       // 8 ints
  int*   route    = (int*)(ws + off);  off += (size_t)NEXP * N_TOK * 4;  // 256 KB
  float* gate_val = (float*)(ws + off); off += (size_t)N_TOK * 2 * 4;    // 64 KB
  float* slot_out = (float*)(ws + off); off += (size_t)N_TOK * 2 * DIM * 4; // 32 MB
  __bf16* xbf     = (__bf16*)(ws + off); off += (size_t)N_TOK * DIM * 2;    // 8 MB
  __bf16* W1t     = (__bf16*)(ws + off); off += (size_t)NEXP * DIM * FFN * 2; // 16 MB
  __bf16* W2t     = (__bf16*)(ws + off); off += (size_t)NEXP * FFN * DIM * 2; // 16 MB

  hipMemsetAsync(cnt, 0, NEXP * sizeof(int), stream);
  cvt_x_kernel<<<(N_TOK * DIM / 8) / 256, 256, 0, stream>>>(x, xbf);
  // W1 [E][D][F] -> W1t [E][F][D]
  transpose_cvt_kernel<<<dim3(FFN / 32, DIM / 32, NEXP), dim3(32, 8), 0, stream>>>(W1, W1t, DIM, FFN);
  // W2 [E][F][D] -> W2t [E][D][F]
  transpose_cvt_kernel<<<dim3(DIM / 32, FFN / 32, NEXP), dim3(32, 8), 0, stream>>>(W2, W2t, FFN, DIM);
  gate_kernel<<<N_TOK / 256, 256, 0, stream>>>(x, Wg, cnt, route, gate_val);
  moe_expert_kernel<<<dim3(N_TOK / TILE_M, NEXP), 256, 0, stream>>>(
      xbf, W1t, b1, W2t, b2, cnt, route, gate_val, slot_out);
  combine_kernel<<<(N_TOK * DIM / 4) / 256, 256, 0, stream>>>(slot_out, out);
}
